// AggOnlyNet_14998025798269
// MI455X (gfx1250) — matrix operand order repaired and checked by reading, not compiled, not run on hardware
//
#include <hip/hip_runtime.h>

// Problem constants (from the reference)
#define NN 50000
#define EE 800000
#define DD 64
#define NHOP 3   // K in reference
#define NLAY 2   // L
#define EPSV 1e-5f

typedef __attribute__((ext_vector_type(16))) _Float16 v16h;
typedef __attribute__((ext_vector_type(8)))  float    v8f;

// ---------------- utility kernels ----------------
__global__ void k_fill_f32(float* p, float v, int n) {
  int i = blockIdx.x * blockDim.x + threadIdx.x;
  if (i < n) p[i] = v;
}
__global__ void k_fill_u32(unsigned* p, unsigned v, int n) {
  int i = blockIdx.x * blockDim.x + threadIdx.x;
  if (i < n) p[i] = v;
}
__global__ void k_copy_f32(float* dst, const float* src, int n) {
  int i = blockIdx.x * blockDim.x + threadIdx.x;
  if (i < n) dst[i] = src[i];
}

// ---------------- graph normalization ----------------
__global__ void k_deg(const int* __restrict__ col, const float* __restrict__ ea,
                      float* __restrict__ deg, int e_cnt) {
  int e = blockIdx.x * blockDim.x + threadIdx.x;
  if (e < e_cnt) atomicAdd(&deg[col[e]], ea[e]);
}
__global__ void k_dinv(float* __restrict__ deg, int n) {
  int i = blockIdx.x * blockDim.x + threadIdx.x;
  if (i < n) {
    float d = deg[i];
    deg[i] = (d > 0.0f) ? rsqrtf(fmaxf(d, EPSV)) : 0.0f;
  }
}
__global__ void k_edgenorm(const int* __restrict__ row, const int* __restrict__ col,
                           const float* __restrict__ ea, const float* __restrict__ dinv,
                           float* __restrict__ nrm, int e_cnt) {
  int e = blockIdx.x * blockDim.x + threadIdx.x;
  if (e < e_cnt) nrm[e] = dinv[row[e]] * ea[e] * dinv[col[e]];
}

// ---------------- sparse propagation: one wave32 per edge, 2 feats/lane -------
__global__ void k_spmm64(const int* __restrict__ row, const int* __restrict__ col,
                         const float* __restrict__ nrm, const float* __restrict__ src,
                         float* __restrict__ dst, int e_cnt) {
  int e = blockIdx.x * 8 + (threadIdx.x >> 5);
  if (e >= e_cnt) return;
  int t = threadIdx.x & 31;
  float w = nrm[e];
  const float* s = src + (size_t)row[e] * DD;
  float*       d = dst + (size_t)col[e] * DD;
  atomicAdd(&d[t],      w * s[t]);
  atomicAdd(&d[t + 32], w * s[t + 32]);
}
__global__ void k_spmm1(const int* __restrict__ row, const int* __restrict__ col,
                        const float* __restrict__ nrm, const float* __restrict__ src,
                        float* __restrict__ dst, int e_cnt) {
  int e = blockIdx.x * blockDim.x + threadIdx.x;
  if (e < e_cnt) atomicAdd(&dst[col[e]], nrm[e] * src[row[e]]);
}

// ---------------- instance norm (column stats over N rows) ----------------
__global__ void k_colstats(const float* __restrict__ h, float* __restrict__ stats,
                           int n, int D) {
  __shared__ float ssum[64];
  __shared__ float ssq[64];
  int tid = threadIdx.x;  // 256
  if (tid < 64) { ssum[tid] = 0.f; ssq[tid] = 0.f; }
  __syncthreads();
  int c = tid % D;
  int rpb = 256 / D;
  int r = blockIdx.x * rpb + tid / D;
  int stride = gridDim.x * rpb;
  float ls = 0.f, lq = 0.f;
  for (; r < n; r += stride) {
    float v = h[r * D + c];
    ls += v; lq += v * v;
  }
  atomicAdd(&ssum[c], ls);
  atomicAdd(&ssq[c], lq);
  __syncthreads();
  if (tid < D) {
    atomicAdd(&stats[tid], ssum[tid]);
    atomicAdd(&stats[64 + tid], ssq[tid]);
  }
}
__global__ void k_instnorm(float* __restrict__ h, const float* __restrict__ stats,
                           int n, int D) {
  int i = blockIdx.x * blockDim.x + threadIdx.x;
  if (i >= n * D) return;
  int c = i % D;
  float inv_n = 1.0f / (float)n;
  float mu = stats[c] * inv_n;
  float var = fmaxf(stats[64 + c] * inv_n - mu * mu, 0.0f);
  h[i] = (h[i] - mu) * rsqrtf(var + EPSV);
}

// ---------------- WMMA GEMM: out(N x 64) = op(A(N x 64) @ W(64 x 64)) ---------
// 8 waves/block; each wave computes one 16-row stripe across all 64 columns.
// W is staged once per block into LDS, transposed and pre-converted to f16, so
// each B operand is a contiguous 16-byte-aligned v16h (ds_load_b128 pairs).
// flags: bit0 = accumulate into existing out, bit1 = add bias, bit2 = relu
__global__ __launch_bounds__(256) void k_gemm64(
    const float* __restrict__ A, const float* __restrict__ W,
    const float* __restrict__ bias, float* __restrict__ out,
    int flags, int nstripes) {
  __shared__ alignas(16) _Float16 Wt[64 * 64];  // Wt[n][k] = W[k][n]
  const int tid = threadIdx.x;
  // cooperative transpose + f32->f16 convert of the 64x64 weight tile
  for (int idx = tid * 4; idx < 4096; idx += 1024) {
    const float4 wv = *(const float4*)(W + idx);
    int k = idx >> 6, n = idx & 63;
    Wt[(n + 0) * 64 + k] = (_Float16)wv.x;
    Wt[(n + 1) * 64 + k] = (_Float16)wv.y;
    Wt[(n + 2) * 64 + k] = (_Float16)wv.z;
    Wt[(n + 3) * 64 + k] = (_Float16)wv.w;
  }
  __syncthreads();

  const int wave = tid >> 5;
  const int lane = tid & 31;
  const int stripe = blockIdx.x * 8 + wave;
  if (stripe >= nstripes) return;
  const int row0 = stripe * 16;
  const int g  = lane >> 4;   // lane-half
  const int mn = lane & 15;   // M for A loads, N for B/C

  v8f c[4];
#pragma unroll
  for (int ct = 0; ct < 4; ++ct) {
    float binit = (flags & 2) ? bias[ct * 16 + mn] : 0.0f;
#pragma unroll
    for (int r = 0; r < 8; ++r) {
      float v = binit;
      if (flags & 1) v += out[(size_t)(row0 + r + 8 * g) * DD + ct * 16 + mn];
      c[ct][r] = v;
    }
  }

#pragma unroll
  for (int k0 = 0; k0 < 64; k0 += 32) {
    // A 16x32 f16 layout: lane holds row M=mn, K = 16*g + e (contiguous)
    v16h a;
#pragma unroll
    for (int e = 0; e < 16; ++e) {
      int ka = e + 16 * g;
      a[e] = (_Float16)A[(size_t)(row0 + mn) * DD + k0 + ka];
    }
    // reuse A across the 4 column tiles
#pragma unroll
    for (int ct = 0; ct < 4; ++ct) {
      // B 32x16 f16 layout: lane holds col N=mn, K = k0 + 16*g + e (contiguous)
      v16h b = *(const v16h*)&Wt[(ct * 16 + mn) * 64 + k0 + 16 * g];
      c[ct] = __builtin_amdgcn_wmma_f32_16x16x32_f16(false, a, false, b,
                                                     (short)0, c[ct], false, false);
    }
  }

#pragma unroll
  for (int ct = 0; ct < 4; ++ct) {
#pragma unroll
    for (int r = 0; r < 8; ++r) {
      float v = c[ct][r];
      if (flags & 4) v = fmaxf(v, 0.0f);
      out[(size_t)(row0 + r + 8 * g) * DD + ct * 16 + mn] = v;
    }
  }
}

// ---------------- block-0 TAGConv epilogue (scalar h, W0: (K+1) x 64) ---------
__global__ void k_tag0(const float* __restrict__ s /* (K+1)*N */,
                       const float* __restrict__ W0, const float* __restrict__ b,
                       float* __restrict__ out, int n) {
  int i = blockIdx.x * blockDim.x + threadIdx.x;
  if (i >= n * DD) return;
  int r = i >> 6, d = i & 63;
  float acc = b[d];
#pragma unroll
  for (int kk = 0; kk <= NHOP; ++kk) acc += s[kk * n + r] * W0[kk * DD + d];
  out[i] = fmaxf(acc, 0.0f);
}

// ---------------- final row dot: scores = relu(h @ wf + bf) -------------------
__global__ void k_rowdot(const float* __restrict__ h, const float* __restrict__ wf,
                         const float* __restrict__ bf, float* __restrict__ sc) {
  __shared__ float red[64];
  int rrow = blockIdx.x;
  int t = threadIdx.x;  // 64
  red[t] = h[(size_t)rrow * DD + t] * wf[t];
  __syncthreads();
  for (int s = 32; s > 0; s >>= 1) {
    if (t < s) red[t] += red[t + s];
    __syncthreads();
  }
  if (t == 0) sc[rrow] = fmaxf(red[0] + bf[0], 0.0f);
}

// ---------------- exact deterministic top-k via 64-bit radix select -----------
__global__ void k_keys(const float* __restrict__ sc, unsigned long long* __restrict__ key,
                       int n) {
  int i = blockIdx.x * blockDim.x + threadIdx.x;
  if (i >= n) return;
  unsigned u = __float_as_uint(sc[i]);
  u = (u & 0x80000000u) ? ~u : (u | 0x80000000u);  // order-preserving map
  // unique composite key; lower index wins ties (matches lax.top_k)
  key[i] = ((unsigned long long)u << 32) | (unsigned)(~(unsigned)i);
}
__global__ void k_selinit(unsigned long long* st, const int* kptr) {
  st[0] = 0ULL;                         // prefix
  st[1] = (unsigned long long)(*kptr);  // remaining rank
}
__global__ void k_hist(const unsigned long long* __restrict__ key,
                       const unsigned long long* __restrict__ st,
                       unsigned* __restrict__ hist, int n, int pass) {
  unsigned long long prefix = st[0];
  int shift = pass * 8;
  for (int i = blockIdx.x * blockDim.x + threadIdx.x; i < n;
       i += gridDim.x * blockDim.x) {
    unsigned long long kk = key[i];
    bool match = (pass == 7) || ((kk >> (shift + 8)) == (prefix >> (shift + 8)));
    if (match) atomicAdd(&hist[(unsigned)((kk >> shift) & 0xFFu)], 1u);
  }
}
__global__ void k_select(unsigned long long* st, const unsigned* __restrict__ hist,
                         int pass) {
  unsigned long long prefix = st[0];
  unsigned rem = (unsigned)st[1];
  int shift = pass * 8;
  for (int b = 255; b >= 0; --b) {
    unsigned c = hist[b];
    if (rem <= c) { prefix |= ((unsigned long long)b) << shift; break; }
    rem -= c;
  }
  st[0] = prefix;
  st[1] = (unsigned long long)rem;
}
__global__ void k_mark(const unsigned long long* __restrict__ key,
                       const unsigned long long* __restrict__ st,
                       float* __restrict__ h, int n) {
  int i = blockIdx.x * blockDim.x + threadIdx.x;
  if (i < n) h[i] = (key[i] >= st[0]) ? 1.0f : 0.0f;
}

// =============================================================================
extern "C" void kernel_launch(void* const* d_in, const int* in_sizes, int n_in,
                              void* d_out, int out_size, void* d_ws, size_t ws_size,
                              hipStream_t stream) {
  const float* x         = (const float*)d_in[0];
  const float* edge_attr = (const float*)d_in[1];
  const float* tag_w0    = (const float*)d_in[2];   // (L, K+1, 1, D)
  const float* tag_b0    = (const float*)d_in[3];   // (L, D)
  const float* tag_w     = (const float*)d_in[4];   // (L, 5, K+1, D, D)
  const float* tag_b     = (const float*)d_in[5];   // (L, 5, D)
  const float* mlp_w     = (const float*)d_in[6];   // (L, 6, 4, D, D)
  const float* mlp_b     = (const float*)d_in[7];   // (L, 6, 4, D)
  const float* mlp_w5    = (const float*)d_in[8];   // (L, 5, D, D)
  const float* mlp_b5    = (const float*)d_in[9];   // (L, 5, D)
  const float* mlp_wf    = (const float*)d_in[10];  // (L, D, 1)
  const float* mlp_bf    = (const float*)d_in[11];  // (L, 1)
  const int*   eidx      = (const int*)d_in[12];    // (2, E)
  const int*   kptr      = (const int*)d_in[13];    // scalar k
  (void)in_sizes; (void)n_in; (void)out_size; (void)ws_size;

  const int* row = eidx;
  const int* col = eidx + EE;

  // ---- carve workspace ----
  char* wsb = (char*)d_ws;
  size_t off = 0;
  auto carve = [&](size_t bytes) -> void* {
    void* p = wsb + off;
    off = (off + bytes + 255) & ~(size_t)255;
    return p;
  };
  float* nrm   = (float*)carve((size_t)EE * 4);
  float* deg   = (float*)carve((size_t)NN * 4);
  float* B0    = (float*)carve((size_t)NN * DD * 4);
  float* B1    = (float*)carve((size_t)NN * DD * 4);
  float* B2    = (float*)carve((size_t)NN * DD * 4);
  float* sbuf  = (float*)carve((size_t)(NHOP + 1) * NN * 4);
  float* sc    = (float*)carve((size_t)NN * 4);
  float* stats = (float*)carve(128 * 4);
  unsigned long long* key = (unsigned long long*)carve((size_t)NN * 8);
  unsigned long long* st  = (unsigned long long*)carve(16);
  unsigned* hist          = (unsigned*)carve(256 * 4);

  float* bufs[3] = {B0, B1, B2};
  auto other1 = [&](const float* a) -> float* {
    for (int i = 0; i < 3; ++i) if (bufs[i] != a) return bufs[i];
    return B0;
  };
  auto pick2 = [&](const float* a, float*& p, float*& q) {
    float* t[2]; int c = 0;
    for (int i = 0; i < 3; ++i) if (bufs[i] != a) t[c++] = bufs[i];
    p = t[0]; q = t[1];
  };

  auto fillf = [&](float* p, float v, int n) {
    k_fill_f32<<<(n + 255) / 256, 256, 0, stream>>>(p, v, n);
  };
  const int NSTRIPES = NN / 16;  // 3125 exactly
  auto gemm = [&](const float* A, const float* W, const float* b, float* out, int flags) {
    k_gemm64<<<(NSTRIPES + 7) / 8, 256, 0, stream>>>(A, W, b, out, flags, NSTRIPES);
  };

  // ---- edge normalization ----
  fillf(deg, 0.0f, NN);
  k_deg<<<(EE + 255) / 256, 256, 0, stream>>>(col, edge_attr, deg, EE);
  k_dinv<<<(NN + 255) / 256, 256, 0, stream>>>(deg, NN);
  k_edgenorm<<<(EE + 255) / 256, 256, 0, stream>>>(row, col, edge_attr, deg, nrm, EE);

  // ---- h = x (N x 1) ----
  float* h = B0;
  k_copy_f32<<<(NN + 255) / 256, 256, 0, stream>>>(h, x, NN);

  for (int l = 0; l < NLAY; ++l) {
    for (int i = 0; i < 6; ++i) {
      int Dh = (i == 0) ? 1 : DD;
      // instance norm (per feature column over all nodes)
      fillf(stats, 0.0f, 128);
      k_colstats<<<256, 256, 0, stream>>>(h, stats, NN, Dh);
      k_instnorm<<<(NN * Dh + 255) / 256, 256, 0, stream>>>(h, stats, NN, Dh);

      if (i == 0) {
        // TAGConv with scalar features
        k_copy_f32<<<(NN + 255) / 256, 256, 0, stream>>>(sbuf, h, NN);
        for (int kk = 1; kk <= NHOP; ++kk) {
          fillf(sbuf + kk * NN, 0.0f, NN);
          k_spmm1<<<(EE + 255) / 256, 256, 0, stream>>>(
              row, col, nrm, sbuf + (kk - 1) * NN, sbuf + kk * NN, EE);
        }
        float* dst = other1(h);
        k_tag0<<<(NN * DD + 255) / 256, 256, 0, stream>>>(
            sbuf, tag_w0 + (size_t)l * (NHOP + 1) * DD, tag_b0 + (size_t)l * DD,
            dst, NN);
        h = dst;  // N x 64, already relu'd
      } else {
        // TAGConv with 64-dim features: out = sum_kk A^kk h @ W[kk] + b, relu
        const float* Wl = tag_w + ((size_t)(l * 5 + (i - 1)) * (NHOP + 1)) * DD * DD;
        const float* bl = tag_b + (size_t)(l * 5 + (i - 1)) * DD;
        float *acc, *tmp;
        pick2(h, acc, tmp);
        float* src = h;
        gemm(src, Wl, nullptr, acc, 0);  // acc = h @ W0
        for (int kk = 1; kk <= NHOP; ++kk) {
          fillf(tmp, 0.0f, NN * DD);
          k_spmm64<<<EE / 8, 256, 0, stream>>>(row, col, nrm, src, tmp, EE);
          int fl = 1 | ((kk == NHOP) ? (2 | 4) : 0);  // acc; last adds bias+relu
          gemm(tmp, Wl + (size_t)kk * DD * DD, bl, acc, fl);
          float* t2 = src; src = tmp; tmp = t2;
        }
        h = acc;
      }

      // MLP: 4 hidden relu layers
      for (int j = 0; j < 4; ++j) {
        const float* Wj = mlp_w + ((size_t)((l * 6 + i) * 4 + j)) * DD * DD;
        const float* bj = mlp_b + ((size_t)((l * 6 + i) * 4 + j)) * DD;
        float* dst = other1(h);
        gemm(h, Wj, bj, dst, 2 | 4);
        h = dst;
      }
      if (i < 5) {
        const float* W5 = mlp_w5 + (size_t)(l * 5 + i) * DD * DD;
        const float* b5 = mlp_b5 + (size_t)(l * 5 + i) * DD;
        float* dst = other1(h);
        gemm(h, W5, b5, dst, 2 | 4);
        h = dst;
      } else {
        // final D -> 1 projection, relu -> scores
        k_rowdot<<<NN, 64, 0, stream>>>(h, mlp_wf + (size_t)l * DD, mlp_bf + l, sc);
        // exact top-k via unique-key radix select (ties -> lower index, like lax.top_k)
        k_keys<<<(NN + 255) / 256, 256, 0, stream>>>(sc, key, NN);
        k_selinit<<<1, 1, 0, stream>>>(st, kptr);
        for (int pass = 7; pass >= 0; --pass) {
          k_fill_u32<<<1, 256, 0, stream>>>(hist, 0u, 256);
          k_hist<<<256, 256, 0, stream>>>(key, st, hist, NN, pass);
          k_select<<<1, 1, 0, stream>>>(st, hist, pass);
        }
        float* dst = other1(h);
        k_mark<<<(NN + 255) / 256, 256, 0, stream>>>(key, st, dst, NN);
        h = dst;  // N x 1 indicator
      }
    }
  }

  // output = h[:, 0] (N floats)
  k_copy_f32<<<(NN + 255) / 256, 256, 0, stream>>>((float*)d_out, h, NN);
}
